// ClusterNet_35356170780708
// MI455X (gfx1250) — compile-verified, matrix operand order checked by reading, not run on hardware
//
#include <hip/hip_runtime.h>
#include <hip/hip_bf16.h>

typedef __attribute__((ext_vector_type(16))) _Float16 v16h;
typedef __attribute__((ext_vector_type(2)))  _Float16 h2;
typedef __attribute__((ext_vector_type(8)))  float    v8f;

#define KNN 6
#define NB    8
#define NPG   16384
#define NN    (NB*NPG)
#define C0PG  2048
#define C0N   (NB*C0PG)
#define C1PG  256
#define C1N   (NB*C1PG)
#define DD    64
#define NLAY  10

__device__ __forceinline__ v8f wmma16(v16h a, v16h b, v8f c) {
  return __builtin_amdgcn_wmma_f32_16x16x32_f16(false, a, false, b, (short)0, c, false, false);
}

// ---- float <-> order-preserving uint encoding (for segment-max via atomicMax) ----
__device__ __forceinline__ unsigned fenc(float f) {
  unsigned u = __float_as_uint(f);
  return (u & 0x80000000u) ? ~u : (u | 0x80000000u);
}
__device__ __forceinline__ float fdec(unsigned e) {
  return (e & 0x80000000u) ? __uint_as_float(e & 0x7fffffffu) : __uint_as_float(~e);
}

// ============================================================================
// Pack layer weight matrices (fp32 row-major [64][64]) into fp16 B-matrix
// register striping: record per (layer, mat, tile{kstep,ntile}, lane) = 16
// halves (32B) so one global_load_b128x2 fills a v16h.
// K index for VGPR-half v, lane l:  2*(v&3) + 16*(v>>2) + 8*(l>=16) (+pair)
// ============================================================================
__global__ void pack_weights(_Float16* __restrict__ wpk,
                             const float* __restrict__ Wsrc,
                             const float* __restrict__ Wdst,
                             const float* __restrict__ Wlin,
                             const float* __restrict__ pW2,
                             const float* __restrict__ aW1,
                             const float* __restrict__ aW2)
{
  int gid = blockIdx.x * blockDim.x + threadIdx.x;
  const int total = NLAY * 6 * 8 * 512;
  if (gid >= total) return;
  int idx = gid;
  int j    = idx & 15;  idx >>= 4;   // half within v16h
  int lane = idx & 31;  idx >>= 5;
  int t    = idx & 7;   idx >>= 3;   // t = kstep*4 + ntile
  int m    = idx % 6;   idx /= 6;
  int l    = idx;
  int v = j >> 1, pr = j & 1;
  int n = (t & 3) * 16 + (lane & 15);
  int k = (t >> 2) * 32 + 2 * (v & 3) + 16 * (v >> 2) + 8 * (lane >> 4) + pr;
  const float* W;
  switch (m) {
    case 0: W = Wsrc; break;
    case 1: W = Wdst; break;
    case 2: W = Wlin; break;
    case 3: W = pW2;  break;
    case 4: W = aW1;  break;
    default: W = aW2; break;
  }
  wpk[(size_t)gid] = (_Float16)W[(size_t)l * 4096 + k * 64 + n];
  // note: gid == ((l*6+m)*8 + t)*512 + lane*16 + j  by construction
}

// ============================================================================
// Fused PointTransformerConv layer.
//  - workgroup: 96 threads = 3 waves, 8 dst nodes = 48 edges = 3 x 16-edge tiles
//  - all 64x64 matmuls via v_wmma_f32_16x16x32_f16
//  - C-layout -> A-layout transposes via LDS staging
//  - per-node softmax(6) + weighted max through LDS
// ============================================================================
__global__ __launch_bounds__(96)
void ptconv_layer(const float* __restrict__ hin, float* __restrict__ hout,
                  const float* __restrict__ pos, const int* __restrict__ src,
                  const _Float16* __restrict__ wpk,
                  const float* __restrict__ pW1, const float* __restrict__ pB1,
                  const float* __restrict__ pB2, const float* __restrict__ aB1,
                  const float* __restrict__ aB2, int l)
{
  __shared__ _Float16 stage[3][16][64];   // per-wave transpose staging
  __shared__ float Als[48][64];           // attention logits per edge
  __shared__ float Vls[48][64];           // value (xv[src]+delta) per edge

  const int tid  = threadIdx.x;
  const int w    = tid >> 5;
  const int lane = tid & 31;
  const int m    = lane & 15;        // edge row within this wave's tile
  const int hi   = lane >> 4;        // lane half (affects K striping)
  const int node0 = blockIdx.x * 8;
  const int e0    = node0 * KNN + w * 16;
  const int e     = e0 + m;
  const int srcn  = src[e];
  const int dstn  = e / KNN;

  const _Float16* WL = wpk + (size_t)l * 6 * 8 * 512;
  auto bld = [&](int mat, int ks, int nt) -> v16h {
    return *(const v16h*)(WL + (((mat * 8) + ks * 4 + nt) * 512 + lane * 16));
  };
  auto loadA = [&](int row, v16h* A) {
    const float2* hp = (const float2*)(hin + (size_t)row * DD);
    for (int ks = 0; ks < 2; ++ks)
      for (int v = 0; v < 8; ++v) {
        int kb = ks * 32 + 2 * (v & 3) + 16 * (v >> 2) + 8 * hi;
        float2 p = hp[kb >> 1];
        A[ks][2 * v]     = (_Float16)p.x;
        A[ks][2 * v + 1] = (_Float16)p.y;
      }
  };
  auto stageC = [&](const v8f* C) {
    for (int nt = 0; nt < 4; ++nt)
      for (int r = 0; r < 8; ++r)
        stage[w][r + 8 * hi][nt * 16 + m] = (_Float16)C[nt][r];
  };
  auto unstageA = [&](v16h* A) {
    const unsigned* sp = (const unsigned*)&stage[w][m][0];
    for (int ks = 0; ks < 2; ++ks)
      for (int v = 0; v < 8; ++v) {
        int kb = ks * 32 + 2 * (v & 3) + 16 * (v >> 2) + 8 * hi;
        h2 p = __builtin_bit_cast(h2, sp[kb >> 1]);
        A[ks][2 * v]     = p.x;
        A[ks][2 * v + 1] = p.y;
      }
  };

  // ---- delta = relu(relu((pos_i - pos_j) @ pW1 + pB1) @ pW2 + pB2) ----
  const float dx = pos[dstn * 2 + 0] - pos[srcn * 2 + 0];
  const float dy = pos[dstn * 2 + 1] - pos[srcn * 2 + 1];
  const float* pw1 = pW1 + (size_t)l * 128;   // [2][64]
  const float* pb1 = pB1 + (size_t)l * 64;
  v16h dA[2];
  for (int ks = 0; ks < 2; ++ks)
    for (int j = 0; j < 16; ++j) {
      int v = j >> 1, pr = j & 1;
      int k = ks * 32 + 2 * (v & 3) + 16 * (v >> 2) + 8 * hi + pr;
      float h1 = dx * pw1[k] + dy * pw1[64 + k] + pb1[k];
      dA[ks][j] = (_Float16)(h1 > 0.f ? h1 : 0.f);
    }
  v8f dC[4];
  for (int nt = 0; nt < 4; ++nt) {
    v8f c = {};
    c = wmma16(dA[0], bld(3, 0, nt), c);
    c = wmma16(dA[1], bld(3, 1, nt), c);
    float b = pB2[l * 64 + nt * 16 + m];
    for (int r = 0; r < 8; ++r) { float x = c[r] + b; c[r] = x > 0.f ? x : 0.f; }
    dC[nt] = c;
  }

  // ---- a_src/xv from h[src], a_dst from h[dst] ----
  v16h A[2];
  loadA(srcn, A);
  v8f tC[4], vC[4];
  for (int nt = 0; nt < 4; ++nt) {          // t = delta - a_src
    v8f c = {};
    c = wmma16(A[0], bld(0, 0, nt), c);
    c = wmma16(A[1], bld(0, 1, nt), c);
    tC[nt] = dC[nt] - c;
  }
  for (int nt = 0; nt < 4; ++nt) {          // v = xv + delta
    v8f c = {};
    c = wmma16(A[0], bld(2, 0, nt), c);
    c = wmma16(A[1], bld(2, 1, nt), c);
    vC[nt] = c + dC[nt];
  }
  loadA(dstn, A);
  for (int nt = 0; nt < 4; ++nt) {          // t += a_dst
    v8f c = {};
    c = wmma16(A[0], bld(1, 0, nt), c);
    c = wmma16(A[1], bld(1, 1, nt), c);
    tC[nt] = tC[nt] + c;
  }

  // ---- attn mlp: a = relu(relu(t@aW1+aB1)@aW2+aB2), with LDS transposes ----
  stageC(tC); __syncthreads(); unstageA(A); __syncthreads();
  v8f aC[4];
  for (int nt = 0; nt < 4; ++nt) {
    v8f c = {};
    c = wmma16(A[0], bld(4, 0, nt), c);
    c = wmma16(A[1], bld(4, 1, nt), c);
    float b = aB1[l * 64 + nt * 16 + m];
    for (int r = 0; r < 8; ++r) { float x = c[r] + b; c[r] = x > 0.f ? x : 0.f; }
    aC[nt] = c;
  }
  stageC(aC); __syncthreads(); unstageA(A); __syncthreads();
  for (int nt = 0; nt < 4; ++nt) {
    v8f c = {};
    c = wmma16(A[0], bld(5, 0, nt), c);
    c = wmma16(A[1], bld(5, 1, nt), c);
    float b = aB2[l * 64 + nt * 16 + m];
    for (int r = 0; r < 8; ++r) { float x = c[r] + b; c[r] = x > 0.f ? x : 0.f; }
    aC[nt] = c;
  }

  // ---- publish per-edge a and v, then softmax(6)+max per (node,feature) ----
  for (int nt = 0; nt < 4; ++nt)
    for (int r = 0; r < 8; ++r) {
      int er = w * 16 + r + 8 * hi;
      Als[er][nt * 16 + m] = aC[nt][r];
      Vls[er][nt * 16 + m] = vC[nt][r];
    }
  __syncthreads();

  for (int it = tid; it < 8 * 64; it += 96) {
    int node = it >> 6, f = it & 63;
    int eb = node * KNN;
    float mx = -__builtin_inff();
    for (int j = 0; j < KNN; ++j) mx = fmaxf(mx, Als[eb + j][f]);
    float ex[KNN]; float s = 0.f;
    for (int j = 0; j < KNN; ++j) { ex[j] = expf(Als[eb + j][f] - mx); s += ex[j]; }
    float o = -__builtin_inff();
    for (int j = 0; j < KNN; ++j) o = fmaxf(o, (ex[j] / s) * Vls[eb + j][f]);
    hout[(size_t)(node0 + node) * DD + f] = o;
  }
}

// ============================================================================
// segment-max pooling via monotonic-uint atomicMax
// ============================================================================
__global__ void init_enc(unsigned* __restrict__ enc, int n) {
  int i = blockIdx.x * blockDim.x + threadIdx.x;
  if (i < n) enc[i] = 0x007FFFFFu;   // fenc(-inf)
}
__global__ void pool_scatter(const float* __restrict__ hin, unsigned* __restrict__ enc,
                             const int* __restrict__ scid, const int* __restrict__ batch,
                             int nper, int cpg, int n)
{
  int i = blockIdx.x * blockDim.x + threadIdx.x;
  if (i >= n * DD) return;
  int node = i >> 6, f = i & 63;
  int b = batch ? batch[node] : (node / nper);
  int c = (scid ? scid[node] : 0) + b * cpg;
  atomicMax(&enc[(size_t)c * DD + f], fenc(hin[(size_t)node * DD + f]));
}
__global__ void decode_enc(float* __restrict__ buf, int n) {
  int i = blockIdx.x * blockDim.x + threadIdx.x;
  if (i < n) {
    unsigned u = ((unsigned*)buf)[i];
    buf[i] = fdec(u);
  }
}

// ============================================================================
// brute-force knn (k=6, with self-loop) per graph; src = neighbor indices
// ============================================================================
__global__ void knn_kernel(const float* __restrict__ pos, int* __restrict__ src,
                           int npg, int total)
{
  int p = blockIdx.x * blockDim.x + threadIdx.x;
  if (p >= total) return;
  int g = p / npg, base = g * npg;
  float px = pos[p * 2], py = pos[p * 2 + 1];
  float bd[KNN]; int bi[KNN];
  for (int j = 0; j < KNN; ++j) { bd[j] = __builtin_inff(); bi[j] = 0; }
  for (int q = 0; q < npg; ++q) {
    float ddx = px - pos[(base + q) * 2];
    float ddy = py - pos[(base + q) * 2 + 1];
    float d2 = ddx * ddx + ddy * ddy;
    int j = KNN;
    while (j > 0 && d2 < bd[j - 1]) --j;
    if (j < KNN) {
      for (int t = KNN - 1; t > j; --t) { bd[t] = bd[t - 1]; bi[t] = bi[t - 1]; }
      bd[j] = d2; bi[j] = q;
    }
  }
  for (int j = 0; j < KNN; ++j) src[p * KNN + j] = base + bi[j];
}

__global__ void head_kernel(const float* __restrict__ g, const float* __restrict__ w,
                            const float* __restrict__ b, float* __restrict__ out)
{
  int t = threadIdx.x;
  if (t >= NB * 2) return;
  int bb = t >> 1, o = t & 1;
  float s = b[o];
  for (int f = 0; f < DD; ++f) s += g[bb * DD + f] * w[f * 2 + o];
  out[t] = s;
}

extern "C" void kernel_launch(void* const* d_in, const int* in_sizes, int n_in,
                              void* d_out, int out_size, void* d_ws, size_t ws_size,
                              hipStream_t stream) {
  const float* x      = (const float*)d_in[0];
  const float* pos_c  = (const float*)d_in[1];
  const int*   src_c  = (const int*)d_in[2];
  const int*   batch  = (const int*)d_in[4];
  const int*   sc_id0 = (const int*)d_in[5];
  const float* pos_s0 = (const float*)d_in[6];
  const int*   sc_id1 = (const int*)d_in[7];
  const float* pos_s1 = (const float*)d_in[8];
  const float* Wlin = (const float*)d_in[9];
  const float* Wsrc = (const float*)d_in[10];
  const float* Wdst = (const float*)d_in[11];
  const float* pW1  = (const float*)d_in[12];
  const float* pB1  = (const float*)d_in[13];
  const float* pW2  = (const float*)d_in[14];
  const float* pB2  = (const float*)d_in[15];
  const float* aW1  = (const float*)d_in[16];
  const float* aB1  = (const float*)d_in[17];
  const float* aW2  = (const float*)d_in[18];
  const float* aB2  = (const float*)d_in[19];
  const float* head_w = (const float*)d_in[20];
  const float* head_b = (const float*)d_in[21];

  // workspace layout
  float*     hA   = (float*)d_ws;                         // NN*64
  float*     hB   = hA + (size_t)NN * DD;                 // NN*64
  _Float16*  wpk  = (_Float16*)(hB + (size_t)NN * DD);    // 10*6*8*512 halves
  int*       sE0  = (int*)(wpk + NLAY * 6 * 8 * 512);     // C0N*6
  int*       sE1  = sE0 + C0N * KNN;                      // C1N*6
  float*     gbuf = (float*)(sE1 + C1N * KNN);            // NB*64

  // 1) pack weights into WMMA B-layout (fp16)
  pack_weights<<<(NLAY * 6 * 8 * 512 + 255) / 256, 256, 0, stream>>>(
      wpk, Wsrc, Wdst, Wlin, pW2, aW1, aW2);

  // 2) cluster-level layers 0..3 (N nodes, given knn edges)
  ptconv_layer<<<NN / 8, 96, 0, stream>>>(x,  hA, pos_c, src_c, wpk, pW1, pB1, pB2, aB1, aB2, 0);
  ptconv_layer<<<NN / 8, 96, 0, stream>>>(hA, hB, pos_c, src_c, wpk, pW1, pB1, pB2, aB1, aB2, 1);
  ptconv_layer<<<NN / 8, 96, 0, stream>>>(hB, hA, pos_c, src_c, wpk, pW1, pB1, pB2, aB1, aB2, 2);
  ptconv_layer<<<NN / 8, 96, 0, stream>>>(hA, hB, pos_c, src_c, wpk, pW1, pB1, pB2, aB1, aB2, 3);

  // 3) pool clusters -> SC0 (segment max)
  init_enc<<<(C0N * DD + 255) / 256, 256, 0, stream>>>((unsigned*)hA, C0N * DD);
  pool_scatter<<<(NN * DD + 255) / 256, 256, 0, stream>>>(hB, (unsigned*)hA, sc_id0, batch, NPG, C0PG, NN);
  decode_enc<<<(C0N * DD + 255) / 256, 256, 0, stream>>>(hA, C0N * DD);

  // 4) knn on SC0 + layers 4..6
  knn_kernel<<<(C0N + 127) / 128, 128, 0, stream>>>(pos_s0, sE0, C0PG, C0N);
  ptconv_layer<<<C0N / 8, 96, 0, stream>>>(hA, hB, pos_s0, sE0, wpk, pW1, pB1, pB2, aB1, aB2, 4);
  ptconv_layer<<<C0N / 8, 96, 0, stream>>>(hB, hA, pos_s0, sE0, wpk, pW1, pB1, pB2, aB1, aB2, 5);
  ptconv_layer<<<C0N / 8, 96, 0, stream>>>(hA, hB, pos_s0, sE0, wpk, pW1, pB1, pB2, aB1, aB2, 6);

  // 5) pool SC0 -> SC1
  init_enc<<<(C1N * DD + 255) / 256, 256, 0, stream>>>((unsigned*)hA, C1N * DD);
  pool_scatter<<<(C0N * DD + 255) / 256, 256, 0, stream>>>(hB, (unsigned*)hA, sc_id1, nullptr, C0PG, C1PG, C0N);
  decode_enc<<<(C1N * DD + 255) / 256, 256, 0, stream>>>(hA, C1N * DD);

  // 6) knn on SC1 + layers 7..9
  knn_kernel<<<(C1N + 127) / 128, 128, 0, stream>>>(pos_s1, sE1, C1PG, C1N);
  ptconv_layer<<<C1N / 8, 96, 0, stream>>>(hA, hB, pos_s1, sE1, wpk, pW1, pB1, pB2, aB1, aB2, 7);
  ptconv_layer<<<C1N / 8, 96, 0, stream>>>(hB, hA, pos_s1, sE1, wpk, pW1, pB1, pB2, aB1, aB2, 8);
  ptconv_layer<<<C1N / 8, 96, 0, stream>>>(hA, hB, pos_s1, sE1, wpk, pW1, pB1, pB2, aB1, aB2, 9);

  // 7) global max pool + linear head
  init_enc<<<(NB * DD + 255) / 256, 256, 0, stream>>>((unsigned*)gbuf, NB * DD);
  pool_scatter<<<(C1N * DD + 255) / 256, 256, 0, stream>>>(hB, (unsigned*)gbuf, nullptr, nullptr, C1PG, 1, C1N);
  decode_enc<<<(NB * DD + 255) / 256, 256, 0, stream>>>(gbuf, NB * DD);
  head_kernel<<<1, 32, 0, stream>>>(gbuf, head_w, head_b, (float*)d_out);
}
